// PartModel_55748675502102
// MI455X (gfx1250) — matrix operand order repaired and checked by reading, not compiled, not run on hardware
//
#include <hip/hip_runtime.h>
#include <hip/hip_bf16.h>
#include <stdint.h>

// ---------------------------------------------------------------------------
// Problem constants (match reference)
// ---------------------------------------------------------------------------
#define N_E      65536
#define DIM      100
#define SEQ      6
#define OC       50          // conv output channels
#define DPAD     128         // per-channel K padding (100 -> 128)
#define KPAD     (OC * DPAD) // 6400 padded K for fc1 GEMM
#define NSTEP    (KPAD / 32) // 200 wmma k-steps
#define NTILE    4           // 50 -> 64 columns, 4 n-tiles of 16
#define EPB      32          // edges per block (2 m-tiles of 16)
#define CHUNK_O  5           // output channels staged per LDS chunk
#define CHUNK_K  (CHUNK_O * DPAD)   // 640
#define CHUNK_S  (CHUNK_K / 32)     // 20 wmma steps per chunk
#define NCHUNK   (OC / CHUNK_O)     // 10

typedef __attribute__((ext_vector_type(16))) __bf16 v16bf;
typedef __attribute__((ext_vector_type(8)))  float  v8f;

union AFrag { unsigned u[8]; v16bf v; };
union BFrag { uint4 q[2]; v16bf v; };

static __device__ __forceinline__ unsigned short f2bf(float f) {
    unsigned u = __builtin_bit_cast(unsigned, f);
    u = (u + 0x7FFFu + ((u >> 16) & 1u)) >> 16;   // round-to-nearest-even
    return (unsigned short)u;
}

// ---------------------------------------------------------------------------
// Kernel 1: repack fc1_w (50 x 5000 fp32) into bf16 WMMA B-fragments.
// Layout: Wb[((S*4 + t)*32 + lane)*8 + v] : packed bf16x2 for
//   j = t*16 + (lane&15),  K' = S*32 + kkB,  kkB = (lane>=16 ? 16 : 0) + 2v
//   K' -> (o = K'/128, d = K'%128); value = fc1_w[j, o*100+d] (0 if padded).
// Total: 200*4*32*8 dwords = 819200 bytes in d_ws.
// ---------------------------------------------------------------------------
__global__ __launch_bounds__(256) void pack_w1(const float* __restrict__ fc1_w,
                                               unsigned* __restrict__ Wb) {
    int idx = blockIdx.x * 256 + threadIdx.x;           // 204800 total
    if (idx >= NSTEP * NTILE * 32 * 8) return;
    int v    = idx & 7;
    int lane = (idx >> 3) & 31;
    int t    = (idx >> 8) & 3;
    int S    = idx >> 10;

    int kkB = ((lane >> 4) << 4) + 2 * v;               // even
    int kp  = S * 32 + kkB;                             // padded K index
    int o   = kp >> 7;                                  // < 50 always
    int d   = kp & 127;
    int j   = t * 16 + (lane & 15);

    float f0 = 0.f, f1 = 0.f;
    if (j < OC) {
        if (d < DIM)     f0 = fc1_w[j * (OC * DIM) + o * DIM + d];
        if (d + 1 < DIM) f1 = fc1_w[j * (OC * DIM) + o * DIM + d + 1];
    }
    unsigned pack = (unsigned)f2bf(f0) | ((unsigned)f2bf(f1) << 16);
    Wb[idx] = pack;
}

// ---------------------------------------------------------------------------
// Kernel 2: fused gather (async global->LDS) + conv(+bias,ReLU) +
//           fc1 (bf16 WMMA) + fc2 epilogue.
// One block = 32 edges, 256 threads = 8 waves; wave = mtile*4 + ntile.
// ---------------------------------------------------------------------------
__global__ __launch_bounds__(256) void fused_edge(
    const int*   __restrict__ edge_idx,   // [2, E]
    const int*   __restrict__ edge_type,  // [E]
    const float* __restrict__ x,          // [N, 100]
    const float* __restrict__ g,
    const float* __restrict__ xi,
    const float* __restrict__ gi,
    const float* __restrict__ conv_w,     // [50, 6]
    const float* __restrict__ conv_b,     // [50]
    const unsigned* __restrict__ Wb,      // prepacked B fragments
    const float* __restrict__ fc1_b,      // [50]
    const float* __restrict__ fc2_w,      // [50]
    const float* __restrict__ fc2_b,      // [1]
    float* __restrict__ out)              // [E]
{
    extern __shared__ char smem[];
    // Dynamic LDS is the only LDS allocation -> byte offsets into smem are
    // wave-relative DS addresses usable by async-to-LDS instructions.
    float*          sG  = (float*)(smem);                    // 32*6*100 f32   = 76800 B
    unsigned short* sH  = (unsigned short*)(smem + 76800);   // 32*640  bf16   = 40960 B
    float*          sCW = (float*)(smem + 117760);           // 300 f32
    float*          sCB = (float*)(smem + 118960);           // 50  f32
    float*          sP  = (float*)(smem + 119168);           // 8*32*8 f32     = 8192 B

    const int tid    = threadIdx.x;
    const int wave   = tid >> 5;
    const int lane   = tid & 31;
    const int mt     = wave >> 2;       // m-tile (0/1)
    const int t      = wave & 3;        // n-tile (0..3)
    const int nl     = lane & 15;
    const int hi     = lane >> 4;
    const int base_e = blockIdx.x * EPB;

    // --- conv weights to LDS ------------------------------------------------
    if (tid < OC * SEQ) sCW[tid] = conv_w[tid];
    if (tid < OC)       sCB[tid] = conv_b[tid];

    // --- gather 32 edges x 6 rows x 100 floats, async global->LDS ----------
    // GLOBAL_LOAD_ASYNC_TO_LDS_B128: per-lane 16B copy, ASYNCcnt-tracked,
    // no VGPR round-trip. Both sides 16B aligned (rows are 400B, q*16).
    for (int idx = tid; idx < EPB * SEQ * (DIM / 4); idx += 256) {
        int e = idx / 150;
        int r = idx - e * 150;
        int k = r / 25;
        int q = r - k * 25;
        int ge  = base_e + e;
        int row = edge_idx[ge];
        int col = edge_idx[N_E + ge];
        int et  = edge_type[ge];
        const float* src;
        switch (k) {
            case 0:  src = x  + (long)row * DIM; break;
            case 1:  src = g  + (long)et  * DIM; break;
            case 2:  src = g  + (long)col * DIM; break;
            case 3:  src = xi + (long)row * DIM; break;
            case 4:  src = gi + (long)et  * DIM; break;
            default: src = xi + (long)col * DIM; break;
        }
        const float* gaddr   = src + q * 4;                       // 16B aligned
        unsigned     lds_off = (unsigned)((e * (SEQ * DIM) + k * DIM + q * 4)
                                          * (int)sizeof(float)); // byte offset
        asm volatile("global_load_async_to_lds_b128 %0, %1, off"
                     :: "v"(lds_off), "v"(gaddr) : "memory");
    }
    asm volatile("s_wait_asynccnt 0x0" ::: "memory");
    __syncthreads();

    // --- main loop: conv chunk -> LDS bf16, then WMMA GEMM ------------------
    v8f acc = {};                                   // fp32 accumulator, 8 VGPRs
    const uint4* Wb4   = (const uint4*)Wb;
    const int    mrow  = mt * 16 + nl;              // A-matrix row (edge)
    const unsigned* sHrow = (const unsigned*)sH + mrow * (CHUNK_K / 2);

    for (int c = 0; c < NCHUNK; ++c) {
        // conv + bias + ReLU for channels [c*5, c*5+5), padded d 100..127 = 0
        for (int idx = tid; idx < EPB * CHUNK_K; idx += 256) {
            int e  = idx / CHUNK_K;
            int rr = idx - e * CHUNK_K;
            int ol = rr >> 7;
            int d  = rr & 127;
            float h = 0.f;
            if (d < DIM) {
                int o = c * CHUNK_O + ol;
                float a = sCB[o];
                const float* ge = sG + e * (SEQ * DIM) + d;
                const float* w  = sCW + o * SEQ;
#pragma unroll
                for (int k = 0; k < SEQ; ++k) a += ge[k * DIM] * w[k];
                h = a > 0.f ? a : 0.f;
            }
            sH[idx] = f2bf(h);
        }
        __syncthreads();

        // 20 wmma k-steps over this chunk; wave owns (mtile, ntile), full K
        const int S0 = c * CHUNK_S;
        for (int s = 0; s < CHUNK_S; ++s) {
            AFrag af;
            const unsigned* ap = sHrow + s * 16 + hi * 8;   // dwords
#pragma unroll
            for (int v = 0; v < 4; ++v) af.u[v] = ap[v];        // K 0-7 / 16-23
#pragma unroll
            for (int v = 0; v < 4; ++v) af.u[4 + v] = ap[4 + v];// K 8-15 / 24-31

            BFrag bf;
            long bidx = (((long)(S0 + s) * NTILE + t) * 32 + lane) * 2;
            bf.q[0] = Wb4[bidx];
            bf.q[1] = Wb4[bidx + 1];

            acc = __builtin_amdgcn_wmma_f32_16x16x32_bf16(
                      false, af.v, false, bf.v, (short)0, acc, false, false);
        }
        __syncthreads();
    }

    // --- epilogue: + fc1_b, * fc2_w, deterministic serial reduce over j -------
    {
        int n = t * 16 + nl;                        // hidden unit index
        float b1n = (n < OC) ? fc1_b[n] : 0.f;
        float w2n = (n < OC) ? fc2_w[n] : 0.f;
        float* p = sP + (wave * 32 + lane) * 8;
#pragma unroll
        for (int r = 0; r < 8; ++r) p[r] = (acc[r] + b1n) * w2n;
    }
    __syncthreads();

    if (tid < EPB) {
        int e   = tid;
        int mt2 = e >> 4;
        int ml  = e & 15;
        int r   = ml & 7;
        int h2  = ml >> 3;
        float sum = 0.f;
        for (int t2 = 0; t2 < NTILE; ++t2) {
            int wv = mt2 * 4 + t2;
            const float* p = sP + (wv * 32 + h2 * 16) * 8 + r;
#pragma unroll
            for (int j = 0; j < 16; ++j) sum += p[j * 8];
        }
        out[base_e + e] = sum + fc2_b[0];
    }
}

// ---------------------------------------------------------------------------
// Host launcher
// ---------------------------------------------------------------------------
extern "C" void kernel_launch(void* const* d_in, const int* in_sizes, int n_in,
                              void* d_out, int out_size, void* d_ws, size_t ws_size,
                              hipStream_t stream) {
    const int*   edge_idx  = (const int*)  d_in[0];
    const int*   edge_type = (const int*)  d_in[1];
    const float* x         = (const float*)d_in[2];
    const float* g         = (const float*)d_in[3];
    const float* xi        = (const float*)d_in[4];
    const float* gi        = (const float*)d_in[5];
    const float* conv_w    = (const float*)d_in[6];
    const float* conv_b    = (const float*)d_in[7];
    const float* fc1_w     = (const float*)d_in[8];
    const float* fc1_b     = (const float*)d_in[9];
    const float* fc2_w     = (const float*)d_in[10];
    const float* fc2_b     = (const float*)d_in[11];
    float*       out       = (float*)d_out;
    unsigned*    Wb        = (unsigned*)d_ws;   // needs 819200 B of scratch

    // 1) repack fc1_w into bf16 B-fragments (200*4*32*8 = 204800 dwords)
    {
        int total  = NSTEP * NTILE * 32 * 8;
        int blocks = (total + 255) / 256;
        pack_w1<<<blocks, 256, 0, stream>>>(fc1_w, Wb);
    }

    // 2) fused kernel: 2048 blocks x 256 threads, 127360 B dynamic LDS
    {
        int blocks = N_E / EPB;                 // 2048
        size_t lds = 119168 + 8192;             // 127360 bytes
        fused_edge<<<blocks, 256, lds, stream>>>(
            edge_idx, edge_type, x, g, xi, gi,
            conv_w, conv_b, Wb, fc1_b, fc2_w, fc2_b, out);
    }
}
